// CrossAttentionLayer_64768106824403
// MI455X (gfx1250) — matrix operand order repaired and checked by reading, not compiled, not run on hardware
//
#include <hip/hip_runtime.h>
#include <cstdint>
#include <cstddef>

// ---------------------------------------------------------------------------
// CDNA5 / gfx1250 cross-attention layer.
// wave32, WMMA v_wmma_f32_16x16x32_bf16 for all matmuls, fp32 softmax/LN,
// Tensor Data Mover (tensor_load_to_lds) double-buffered GEMM tile staging.
// Memory-bound workload (~400MB compulsory traffic @ 23.3 TB/s ~= 17us floor),
// so bf16 operands + L2-resident weights is the right operating point.
// ---------------------------------------------------------------------------

#define HIDDEN   4096
#define HEADS    32
#define HEAD_DIM 128
#define SLOTS    128
#define BATCH    2
#define SEQ      4096
#define MROWS    (BATCH * SEQ)   // 8192 flattened (b,s) rows

typedef __attribute__((ext_vector_type(16))) __bf16 v16bf;
typedef __attribute__((ext_vector_type(8)))  float  v8f;
typedef __attribute__((ext_vector_type(4)))  __bf16 v4bf;
typedef __attribute__((ext_vector_type(4)))  unsigned int u32x4;
typedef __attribute__((ext_vector_type(8)))  unsigned int u32x8;

union Frag16 { uint4 q[2]; v16bf v; };  // 32B: one 16-bit WMMA A/B operand

static __device__ __forceinline__ v8f zero8() {
  v8f z = {0.f, 0.f, 0.f, 0.f, 0.f, 0.f, 0.f, 0.f};
  return z;
}

// Low 32 bits of a generic pointer to LDS == LDS byte offset (aperture rule).
static __device__ __forceinline__ unsigned lds_off(const void* p) {
  return (unsigned)(unsigned long long)p;
}

// ---------------------------------------------------------------------------
// TDM: issue a 2D tensor_load_to_lds of a (tile_rows x 32 bf16) tile whose
// rows are `row_stride_elems` bf16 apart in global memory, into LDS laid out
// with an 80-byte row pitch (64B row + 16B pad) == __bf16[rows][40].
// D# per CDNA5 ISA:
//   g0: count=1 | lds_addr | global_addr[56:0] | type=2
//   g1: data_size=2B, pad_enable, pad_interval=3 (16 DW = 64B),
//       pad_amount=3 (4 DW = 16B), tile_dim0=32, tile_dim1=rows,
//       tensor_dim0_stride=row_stride; huge tensor dims (tile never OOB).
// ---------------------------------------------------------------------------
static __device__ __forceinline__ void tdm_load_tile(
    unsigned lds_addr, const __bf16* gptr, unsigned rows,
    unsigned long long row_stride_elems) {
  const unsigned long long ga = (unsigned long long)gptr;
  u32x4 g0;
  g0[0] = 1u;                                   // count=1 (valid), user mode
  g0[1] = lds_addr;                             // LDS byte address
  g0[2] = (unsigned)(ga & 0xFFFFFFFFu);         // global_addr[31:0]
  g0[3] = (unsigned)((ga >> 32) & 0x1FFFFFFu)   // global_addr[56:32]
          | (2u << 30);                         // type=2 ("image")
  const unsigned td_lo16 = 0u, td_hi16 = 0x4000u;  // tensor dims = 1<<30
  u32x8 g1;
  g1[0] = (1u << 16)        // data_size: 2 bytes
        | (1u << 20)        // pad_enable
        | (3u << 22)        // pad_interval: 16 DWORDs (after each 64B row)
        | (3u << 25);       // pad_amount: 4 DWORDs (16B) -> 80B row pitch
  g1[1] = (td_lo16 << 16);                  // atomic_addr=0, tensor_dim0[15:0]
  g1[2] = td_hi16 | (td_lo16 << 16);        // tensor_dim0[31:16], dim1[15:0]
  g1[3] = td_hi16 | (32u << 16);            // tensor_dim1[31:16], tile_dim0=32
  g1[4] = rows;                             // tile_dim1 = rows, tile_dim2 = 0
  g1[5] = (unsigned)row_stride_elems;       // tensor_dim0_stride[31:0]
  g1[6] = (unsigned)(row_stride_elems >> 32);  // stride[47:32], dim1_stride lo=0
  g1[7] = 0u;                               // tensor_dim1_stride hi (unused 2D)
  asm volatile("tensor_load_to_lds %0, %1" : : "s"(g0), "s"(g1) : "memory");
}

// ---------------------------------------------------------------------------
// Kernel 1: fused LayerNorm + fp32 -> bf16 cast. One row per block.
// ---------------------------------------------------------------------------
__global__ __launch_bounds__(256) void ln_cast_kernel(
    const float* __restrict__ x, const float* __restrict__ gamma,
    const float* __restrict__ beta, __bf16* __restrict__ out) {
  const int row = blockIdx.x;
  const float* xr = x + (size_t)row * HIDDEN;
  const int base0 = threadIdx.x * 16;

  float4 v[4];
  float s = 0.f, s2 = 0.f;
#pragma unroll
  for (int i = 0; i < 4; ++i) {
    v[i] = *(const float4*)&xr[base0 + i * 4];
    s  += v[i].x + v[i].y + v[i].z + v[i].w;
    s2 += v[i].x * v[i].x + v[i].y * v[i].y + v[i].z * v[i].z + v[i].w * v[i].w;
  }
#pragma unroll
  for (int m = 1; m < 32; m <<= 1) {
    s  += __shfl_xor(s,  m, 32);
    s2 += __shfl_xor(s2, m, 32);
  }
  __shared__ float sh1[8], sh2[8];
  const int wv = threadIdx.x >> 5;
  if ((threadIdx.x & 31) == 0) { sh1[wv] = s; sh2[wv] = s2; }
  __syncthreads();
  float S = 0.f, S2 = 0.f;
#pragma unroll
  for (int i = 0; i < 8; ++i) { S += sh1[i]; S2 += sh2[i]; }

  const float mu  = S * (1.0f / HIDDEN);
  const float var = S2 * (1.0f / HIDDEN) - mu * mu;
  const float rs  = rsqrtf(var + 1e-5f);

  __bf16* orow = out + (size_t)row * HIDDEN;
#pragma unroll
  for (int i = 0; i < 4; ++i) {
    const int c = base0 + i * 4;
    const float4 g = *(const float4*)&gamma[c];
    const float4 b = *(const float4*)&beta[c];
    v4bf o;
    o[0] = (__bf16)((v[i].x - mu) * rs * g.x + b.x);
    o[1] = (__bf16)((v[i].y - mu) * rs * g.y + b.y);
    o[2] = (__bf16)((v[i].z - mu) * rs * g.z + b.z);
    o[3] = (__bf16)((v[i].w - mu) * rs * g.w + b.w);
    *(v4bf*)&orow[c] = o;
  }
}

// ---------------------------------------------------------------------------
// Kernel 2: elementwise fp32 -> bf16 (weights, memory keys).
// ---------------------------------------------------------------------------
__global__ void cast_f32_bf16_kernel(const float* __restrict__ src,
                                     __bf16* __restrict__ dst, int n4) {
  const int i = blockIdx.x * blockDim.x + threadIdx.x;
  if (i < n4) {
    const float4 v = *(const float4*)&src[(size_t)i * 4];
    v4bf o;
    o[0] = (__bf16)v.x; o[1] = (__bf16)v.y; o[2] = (__bf16)v.z; o[3] = (__bf16)v.w;
    *(v4bf*)&dst[(size_t)i * 4] = o;
  }
}

// ---------------------------------------------------------------------------
// Kernel 3: V -> Vt[b][h][d][t] (bf16) so the P@V B-operand is K(t)-contiguous.
// ---------------------------------------------------------------------------
__global__ void vtranspose_kernel(const float* __restrict__ V,
                                  __bf16* __restrict__ Vt) {
  const int h = blockIdx.x, b = blockIdx.y;
  for (int idx = threadIdx.x; idx < SLOTS * HEAD_DIM; idx += blockDim.x) {
    const int t = idx >> 7;    // slot   (coalesced read along d)
    const int d = idx & 127;   // head dim
    Vt[(((size_t)b * HEADS + h) * HEAD_DIM + d) * SLOTS + t] =
        (__bf16)V[((size_t)b * SLOTS + t) * HIDDEN + h * HEAD_DIM + d];
  }
}

// ---------------------------------------------------------------------------
// Kernel 4: bf16 GEMM, out = A(MxK) * Bt(NxK)^T.  128x128 block tile, 8 waves,
// each wave 64x32 (4x2 WMMA 16x16 tiles). K tiles (128x32) streamed into LDS
// by the Tensor Data Mover, double-buffered: issue k+32 into the alternate
// buffer, s_wait_tensorcnt 2 retires the current buffer's two loads (TENSORcnt
// completes in-order), barrier publishes LDS to all waves.
// ADD_RESIDUAL: fp32 out + residual; else bf16 out.
// ---------------------------------------------------------------------------
template <bool ADD_RESIDUAL>
__global__ __launch_bounds__(256) void gemm_bf16_kernel(
    const __bf16* __restrict__ A, const __bf16* __restrict__ Bt,
    const float* __restrict__ resid, float* __restrict__ out32,
    __bf16* __restrict__ out16, int M, int N, int K) {
  __shared__ __align__(16) __bf16 As[2][128][40];  // 80B pitch == TDM pad
  __shared__ __align__(16) __bf16 Bs[2][128][40];

  const int tid  = threadIdx.x;
  const int lane = tid & 31, wv = tid >> 5;
  const int wm   = wv >> 2;       // 0..1 : 64-row slice
  const int wn   = wv & 3;        // 0..3 : 32-col slice
  const int n16  = lane & 15, half = lane >> 4;
  const int mbase = blockIdx.y * 128;
  const int nbase = blockIdx.x * 128;

  v8f acc[4][2];
#pragma unroll
  for (int i = 0; i < 4; ++i)
#pragma unroll
    for (int j = 0; j < 2; ++j) acc[i][j] = zero8();

  const __bf16* Ab = A  + (size_t)mbase * K;
  const __bf16* Bb = Bt + (size_t)nbase * K;
  const int KT = K / 32;

  if (tid == 0) {  // TDM ignores EXEC; TENSORcnt is per-wave (wave 0 only)
    tdm_load_tile(lds_off(&As[0][0][0]), Ab, 128, (unsigned long long)K);
    tdm_load_tile(lds_off(&Bs[0][0][0]), Bb, 128, (unsigned long long)K);
  }

  int buf = 0;
  for (int kt = 0; kt < KT; ++kt) {
    if (tid == 0) {
      if (kt + 1 < KT) {
        // Prefetch next K tile into the other buffer (all waves finished
        // reading it at the barrier that ended the previous iteration).
        tdm_load_tile(lds_off(&As[buf ^ 1][0][0]), Ab + (kt + 1) * 32, 128,
                      (unsigned long long)K);
        tdm_load_tile(lds_off(&Bs[buf ^ 1][0][0]), Bb + (kt + 1) * 32, 128,
                      (unsigned long long)K);
        __builtin_amdgcn_s_wait_tensorcnt((short)2);  // current buf's 2 done
      } else {
        __builtin_amdgcn_s_wait_tensorcnt((short)0);
      }
    }
    __syncthreads();

    // A frag (16x32, ISA layout): lane m=lane&15; K chunks [h*8,+8), [16+h*8,+8)
    Frag16 af[4];
#pragma unroll
    for (int i = 0; i < 4; ++i) {
      const int r = wm * 64 + i * 16 + n16;
      af[i].q[0] = *(const uint4*)&As[buf][r][half * 8];
      af[i].q[1] = *(const uint4*)&As[buf][r][16 + half * 8];
    }
    // B frag (32x16): lane n=lane&15; K chunks [h*8,+8), [16+h*8,+8)
    Frag16 bf[2];
#pragma unroll
    for (int j = 0; j < 2; ++j) {
      const int r = wn * 32 + j * 16 + n16;
      bf[j].q[0] = *(const uint4*)&Bs[buf][r][half * 8];
      bf[j].q[1] = *(const uint4*)&Bs[buf][r][16 + half * 8];
    }
#pragma unroll
    for (int i = 0; i < 4; ++i)
#pragma unroll
      for (int j = 0; j < 2; ++j)
        acc[i][j] = __builtin_amdgcn_wmma_f32_16x16x32_bf16(
            false, af[i].v, false, bf[j].v, (short)0, acc[i][j], false, false);
    __syncthreads();  // all waves done with buf before TDM refills it
    buf ^= 1;
  }

  // D layout: vgpr r -> (M = half*8 + r, N = lane&15)
#pragma unroll
  for (int i = 0; i < 4; ++i) {
    const int mrow = mbase + wm * 64 + i * 16 + half * 8;
#pragma unroll
    for (int j = 0; j < 2; ++j) {
      const int ncol = nbase + wn * 32 + j * 16 + n16;
#pragma unroll
      for (int r = 0; r < 8; ++r) {
        const size_t idx = (size_t)(mrow + r) * N + ncol;
        if (ADD_RESIDUAL) out32[idx] = resid[idx] + acc[i][j][r];
        else              out16[idx] = (__bf16)acc[i][j][r];
      }
    }
  }
}

// ---------------------------------------------------------------------------
// Kernel 5: cross-attention. Each wave: 16 query rows of one (b,h).
// scores(16x128) = Q(16x128) @ K^T via 8x4 WMMA, masked fp32 softmax
// (per-lane + 16-lane shfl_xor reductions), P staged via per-wave LDS into
// A-operand layout, attn(16x128) = P @ V via 8x4 WMMA against Vt.
// ---------------------------------------------------------------------------
__global__ __launch_bounds__(256) void attn_kernel(
    const __bf16* __restrict__ Q, const __bf16* __restrict__ Kb,
    const __bf16* __restrict__ Vt, const unsigned char* __restrict__ mask,
    __bf16* __restrict__ Aout) {
  __shared__ __align__(16) __bf16 Plds[8][16][136];  // per-wave 16x128 (+8 pad)

  const int stile = blockIdx.x, h = blockIdx.y, b = blockIdx.z;
  const int lane = threadIdx.x & 31, wv = threadIdx.x >> 5;
  const int n16 = lane & 15, half = lane >> 4;
  const int srow = stile * 128 + wv * 16;

  // Q fragments (A operand), K-contiguous in head_dim
  Frag16 qf[4];
  {
    const __bf16* qp = Q + (size_t)(b * SEQ + srow + n16) * HIDDEN + h * HEAD_DIM;
#pragma unroll
    for (int ks = 0; ks < 4; ++ks) {
      qf[ks].q[0] = *(const uint4*)(qp + ks * 32 + half * 16);
      qf[ks].q[1] = *(const uint4*)(qp + ks * 32 + half * 16 + 8);
    }
  }

  // scores: 8 tiles of 16x16 over the 128 slots
  v8f sc[8];
#pragma unroll
  for (int j = 0; j < 8; ++j) {
    v8f acc = zero8();
    const __bf16* kp = Kb + (size_t)(b * SLOTS + 16 * j + n16) * HIDDEN +
                       h * HEAD_DIM + half * 16;
#pragma unroll
    for (int ks = 0; ks < 4; ++ks) {
      Frag16 kf;
      kf.q[0] = *(const uint4*)(kp + ks * 32);
      kf.q[1] = *(const uint4*)(kp + ks * 32 + 8);
      acc = __builtin_amdgcn_wmma_f32_16x16x32_bf16(
          false, qf[ks].v, false, kf.v, (short)0, acc, false, false);
    }
    sc[j] = acc;
  }

  const float scale = 0.088388347648318447f;  // 1/sqrt(128)
  float mcol[8];
#pragma unroll
  for (int j = 0; j < 8; ++j)
    mcol[j] = mask[b * SLOTS + 16 * j + n16] ? 0.0f : -1e30f;
#pragma unroll
  for (int j = 0; j < 8; ++j)
#pragma unroll
    for (int r = 0; r < 8; ++r)
      sc[j][r] = sc[j][r] * scale + mcol[j];

  // softmax per row; D layout: vgpr r holds row (half*8 + r), column 16j+n16;
  // lanes 0-15 / 16-31 hold disjoint rows, so xor-masks {1,2,4,8} reduce each
  // half-wave's 16 columns independently.
#pragma unroll
  for (int r = 0; r < 8; ++r) {
    float mx = sc[0][r];
#pragma unroll
    for (int j = 1; j < 8; ++j) mx = fmaxf(mx, sc[j][r]);
#pragma unroll
    for (int m = 1; m <= 8; m <<= 1) mx = fmaxf(mx, __shfl_xor(mx, m, 32));
    float e[8], s = 0.f;
#pragma unroll
    for (int j = 0; j < 8; ++j) { e[j] = __expf(sc[j][r] - mx); s += e[j]; }
#pragma unroll
    for (int m = 1; m <= 8; m <<= 1) s += __shfl_xor(s, m, 32);
    const float inv = 1.0f / s;
    const int prow = half * 8 + r;
#pragma unroll
    for (int j = 0; j < 8; ++j)
      Plds[wv][prow][16 * j + n16] = (__bf16)(e[j] * inv);
  }
  // per-wave LDS: hardware keeps DS ops in order within a wave; fence the
  // compiler so the A-fragment reads stay below the stores.
  asm volatile("" ::: "memory");

  // P fragments (A operand) from LDS
  Frag16 pf[4];
#pragma unroll
  for (int ks = 0; ks < 4; ++ks) {
    pf[ks].q[0] = *(const uint4*)&Plds[wv][n16][ks * 32 + half * 16];
    pf[ks].q[1] = *(const uint4*)&Plds[wv][n16][ks * 32 + half * 16 + 8];
  }

  // attn = P @ V ; Vt[b][h][d][t] gives K(t)-contiguous B fragments
#pragma unroll
  for (int jd = 0; jd < 8; ++jd) {
    v8f acc = zero8();
    const __bf16* vp = Vt + ((size_t)(b * HEADS + h) * HEAD_DIM + 16 * jd + n16) * SLOTS +
                       half * 16;
#pragma unroll
    for (int ks = 0; ks < 4; ++ks) {
      Frag16 vf;
      vf.q[0] = *(const uint4*)(vp + ks * 32);
      vf.q[1] = *(const uint4*)(vp + ks * 32 + 8);
      acc = __builtin_amdgcn_wmma_f32_16x16x32_bf16(
          false, pf[ks].v, false, vf.v, (short)0, acc, false, false);
    }
#pragma unroll
    for (int r = 0; r < 8; ++r)
      Aout[(size_t)(b * SEQ + srow + half * 8 + r) * HIDDEN + h * HEAD_DIM +
           16 * jd + n16] = (__bf16)acc[r];
  }
}

// ---------------------------------------------------------------------------
// Host launcher. Workspace layout (196 MB total); attention output aliases
// the LN buffer (dead after the Q GEMM).
// ---------------------------------------------------------------------------
extern "C" void kernel_launch(void* const* d_in, const int* in_sizes, int n_in,
                              void* d_out, int out_size, void* d_ws, size_t ws_size,
                              hipStream_t stream) {
  (void)in_sizes; (void)n_in; (void)out_size; (void)ws_size;

  const float* hidden = (const float*)d_in[0];
  const float* mkeys  = (const float*)d_in[1];
  const float* mvals  = (const float*)d_in[2];
  const unsigned char* amask = (const unsigned char*)d_in[3];
  const float* Wq     = (const float*)d_in[4];
  const float* Wout   = (const float*)d_in[5];
  const float* gamma  = (const float*)d_in[6];
  const float* beta   = (const float*)d_in[7];
  float* out = (float*)d_out;

  char* ws = (char*)d_ws;
  const size_t MK  = (size_t)MROWS * HIDDEN;          // 33,554,432 elems
  const size_t WW  = (size_t)HIDDEN * HIDDEN;         // 16,777,216 elems
  const size_t KVN = (size_t)BATCH * SLOTS * HIDDEN;  //  1,048,576 elems
  size_t off = 0;
  __bf16* Xln = (__bf16*)(ws + off); off += MK * 2;   // 64 MB (reused as attn out)
  __bf16* Qb  = (__bf16*)(ws + off); off += MK * 2;   // 64 MB
  __bf16* WqB = (__bf16*)(ws + off); off += WW * 2;   // 32 MB
  __bf16* WoB = (__bf16*)(ws + off); off += WW * 2;   // 32 MB
  __bf16* KbB = (__bf16*)(ws + off); off += KVN * 2;  //  2 MB
  __bf16* VtB = (__bf16*)(ws + off); off += KVN * 2;  //  2 MB

  // 1) LayerNorm + cast
  ln_cast_kernel<<<MROWS, 256, 0, stream>>>(hidden, gamma, beta, Xln);
  // 2) weight / key casts + V transpose
  cast_f32_bf16_kernel<<<(int)(WW / 4 / 256), 256, 0, stream>>>(Wq, WqB, (int)(WW / 4));
  cast_f32_bf16_kernel<<<(int)(WW / 4 / 256), 256, 0, stream>>>(Wout, WoB, (int)(WW / 4));
  cast_f32_bf16_kernel<<<(int)(KVN / 4 / 256), 256, 0, stream>>>(mkeys, KbB, (int)(KVN / 4));
  vtranspose_kernel<<<dim3(HEADS, BATCH), 256, 0, stream>>>(mvals, VtB);
  // 3) Q = LN(x) @ Wq^T   (TDM-staged GEMM)
  gemm_bf16_kernel<false><<<dim3(HIDDEN / 128, MROWS / 128), 256, 0, stream>>>(
      Xln, WqB, nullptr, nullptr, Qb, MROWS, HIDDEN, HIDDEN);
  // 4) attention (writes into Xln region, reused as attn output)
  attn_kernel<<<dim3(SEQ / 128, HEADS, BATCH), 256, 0, stream>>>(
      Qb, KbB, VtB, amask, Xln);
  // 5) out = attn @ Wout^T + residual (fp32, TDM-staged GEMM)
  gemm_bf16_kernel<true><<<dim3(HIDDEN / 128, MROWS / 128), 256, 0, stream>>>(
      Xln, WoB, hidden, out, nullptr, MROWS, HIDDEN, HIDDEN);
}
